// PointnetSAMSG_52750788329674
// MI455X (gfx1250) — compile-verified, matrix operand order checked by reading, not run on hardware
//
#include <hip/hip_runtime.h>
#include <hip/hip_bf16.h>

typedef __bf16 bf16_t;
typedef __attribute__((ext_vector_type(16))) __bf16 v16bf;
typedef __attribute__((ext_vector_type(8)))  __bf16 v8bf;
typedef __attribute__((ext_vector_type(8)))  float  v8f;

#define B_ 16
#define N_ 4096
#define S_ 1024
#define CPTS 16
#define FEAT_TOT 320

// d_out layout (element offsets): new_xyz | new_points_concat | new_idx(int bits)
#define OUT_XYZ  0
#define OUT_FEAT (B_*S_*3)                      // 49152
#define OUT_IDX  (OUT_FEAT + B_*S_*FEAT_TOT)    // 5292032

// workspace layout (byte offsets)
#define IDX0_OFF 0u
#define IDX1_OFF (IDX0_OFF + (size_t)B_*S_*16*4)   // 1 MiB
#define IDX2_OFF (IDX1_OFF + (size_t)B_*S_*32*4)   // +2 MiB
#define WT_OFF   (IDX2_OFF + (size_t)B_*S_*64*4)   // +4 MiB, bf16 weights after

// ---------------------------------------------------------------------------
// Kernel 1: farthest point sampling, one block per batch
// ---------------------------------------------------------------------------
__global__ __launch_bounds__(256)
void fps_kernel(const float* __restrict__ xyz, float* out) {
  const int b   = blockIdx.x;
  const int tid = threadIdx.x;
  __shared__ float dist[N_];
  __shared__ float rv[256];
  __shared__ int   ri[256];
  __shared__ float lastP[3];
  float* nxz  = out + OUT_XYZ;
  int*   nidx = (int*)out + OUT_IDX;

  for (int j = tid; j < N_; j += 256) dist[j] = 1e10f;
  if (tid == 0) {
    nidx[b*S_ + 0] = 0;
    float x = xyz[(size_t)(b*N_)*3 + 0];
    float y = xyz[(size_t)(b*N_)*3 + 1];
    float z = xyz[(size_t)(b*N_)*3 + 2];
    lastP[0] = x; lastP[1] = y; lastP[2] = z;
    nxz[(size_t)(b*S_)*3 + 0] = x;
    nxz[(size_t)(b*S_)*3 + 1] = y;
    nxz[(size_t)(b*S_)*3 + 2] = z;
  }
  __syncthreads();

  for (int it = 1; it < S_; ++it) {
    const float lx = lastP[0], ly = lastP[1], lz = lastP[2];
    float bmax = -1.0f; int bi = 0;
    for (int j = tid; j < N_; j += 256) {
      const float* p = xyz + (size_t)(b*N_ + j)*3;
      float dx = p[0]-lx, dy = p[1]-ly, dz = p[2]-lz;
      float d  = dx*dx + dy*dy + dz*dz;
      d = fminf(dist[j], d);
      dist[j] = d;
      if (d > bmax) { bmax = d; bi = j; }  // first max (ascending j per thread)
    }
    rv[tid] = bmax; ri[tid] = bi;
    __syncthreads();
    for (int st = 128; st > 0; st >>= 1) {
      if (tid < st) {
        float v2 = rv[tid+st]; int i2 = ri[tid+st];
        if (v2 > rv[tid] || (v2 == rv[tid] && i2 < ri[tid])) { rv[tid] = v2; ri[tid] = i2; }
      }
      __syncthreads();
    }
    if (tid == 0) {
      int best = ri[0];
      nidx[b*S_ + it] = best;
      const float* p = xyz + (size_t)(b*N_ + best)*3;
      lastP[0] = p[0]; lastP[1] = p[1]; lastP[2] = p[2];
      nxz[(size_t)(b*S_ + it)*3 + 0] = p[0];
      nxz[(size_t)(b*S_ + it)*3 + 1] = p[1];
      nxz[(size_t)(b*S_ + it)*3 + 2] = p[2];
    }
    __syncthreads();
  }
}

// ---------------------------------------------------------------------------
// Kernel 2: ball query for all 3 radii; one thread per query, xyz staged in LDS
// ---------------------------------------------------------------------------
__global__ __launch_bounds__(256)
void qball_kernel(const float* __restrict__ xyz, const float* out,
                  int* __restrict__ i0, int* __restrict__ i1, int* __restrict__ i2) {
  const int b = blockIdx.x >> 2;
  const int s = ((blockIdx.x & 3) << 8) + threadIdx.x;
  __shared__ float sx[N_], sy[N_], sz[N_];
  for (int j = threadIdx.x; j < N_; j += 256) {
    const float* p = xyz + (size_t)(b*N_ + j)*3;
    sx[j] = p[0]; sy[j] = p[1]; sz[j] = p[2];
  }
  __syncthreads();

  const float* q = out + OUT_XYZ + (size_t)(b*S_ + s)*3;
  const float qx = q[0], qy = q[1], qz = q[2];
  int* o0 = i0 + (size_t)(b*S_ + s)*16;
  int* o1 = i1 + (size_t)(b*S_ + s)*32;
  int* o2 = i2 + (size_t)(b*S_ + s)*64;
  const float r0 = 0.1f*0.1f, r1 = 0.2f*0.2f, r2 = 0.4f*0.4f;
  int c0 = 0, c1 = 0, c2 = 0, f0 = 0, f1 = 0, f2 = 0;

  for (int j = 0; j < N_; ++j) {
    float dx = sx[j]-qx, dy = sy[j]-qy, dz = sz[j]-qz;
    float d2 = dx*dx + dy*dy + dz*dz;
    if (d2 < r2) {
      if (c2 == 0) f2 = j;
      if (c2 < 64) o2[c2++] = j;
      if (d2 < r1) {
        if (c1 == 0) f1 = j;
        if (c1 < 32) o1[c1++] = j;
        if (d2 < r0) {
          if (c0 == 0) f0 = j;
          if (c0 < 16) o0[c0++] = j;
        }
      }
      if (c0 == 16 && c1 == 32 && c2 == 64) break;
    }
  }
  for (int k = c0; k < 16; ++k) o0[k] = f0;  // pad with first in-ball index
  for (int k = c1; k < 32; ++k) o1[k] = f1;
  for (int k = c2; k < 64; ++k) o2[k] = f2;
}

// ---------------------------------------------------------------------------
// Kernel 3: weight preprocessing: f32 [cin][F] -> bf16 transposed padded [F][CPAD]
// ---------------------------------------------------------------------------
__global__ __launch_bounds__(256)
void prep_w_kernel(const float* __restrict__ w, bf16_t* __restrict__ o,
                   int cin, int F, int CPAD) {
  int i = blockIdx.x * 256 + threadIdx.x;
  if (i >= F * CPAD) return;
  int f = i / CPAD, c = i % CPAD;
  o[i] = (c < cin) ? (bf16_t)w[c * F + f] : (bf16_t)0.0f;
}

// ---------------------------------------------------------------------------
// WMMA fragment loaders (16-bit A/B layout: lanes<16 hold K 0..7 / 16..23,
// lanes>=16 hold K 8..15 / 24..31; one row(A)/col(B) per lane mod 16)
// ---------------------------------------------------------------------------
__device__ __forceinline__ v16bf ld_frag_lds(const bf16_t* base, int kt, int lane) {
  const int m  = lane & 15;
  const int kb = (lane >> 4) << 3;
  const bf16_t* p = base + m*128 + kt*32 + kb;
  union { v16bf v; v8bf h[2]; } u;
  u.h[0] = *(const v8bf*)(p);
  u.h[1] = *(const v8bf*)(p + 16);
  return u.v;
}

template<int CPAD>
__device__ __forceinline__ v16bf ld_frag_w(const bf16_t* __restrict__ wt,
                                           int nt, int kt, int lane) {
  const int n  = lane & 15;
  const int kb = (lane >> 4) << 3;
  const bf16_t* p = wt + (size_t)(nt*16 + n)*CPAD + kt*32 + kb;
  union { v16bf v; v8bf h[2]; } u;
  u.h[0] = *(const v8bf*)(p);      // global_load_b128
  u.h[1] = *(const v8bf*)(p + 16);
  return u.v;
}

// one MLP layer on a 16-row tile: LDS(bf16,stride128) x W -> relu -> LDS
template<int CIN, int FOUT>
__device__ __forceinline__ void mlp_layer(const bf16_t* in, const bf16_t* __restrict__ wt,
                                          bf16_t* outb, int lane) {
  const int n  = lane & 15;
  const int rb = (lane >> 4) << 3;   // C layout: VGPR r -> row r + 8*(lane>=16)
  #pragma unroll
  for (int nt = 0; nt < FOUT/16; ++nt) {
    v8f c;
    #pragma unroll
    for (int r = 0; r < 8; ++r) c[r] = 0.0f;
    #pragma unroll
    for (int kt = 0; kt < CIN/32; ++kt) {
      v16bf a = ld_frag_lds(in, kt, lane);
      v16bf w = ld_frag_w<CIN>(wt, nt, kt, lane);
      c = __builtin_amdgcn_wmma_f32_16x16x32_bf16(false, a, false, w, (short)0, c, false, false);
    }
    #pragma unroll
    for (int r = 0; r < 8; ++r)
      outb[(rb + r)*128 + nt*16 + n] = (bf16_t)fmaxf(c[r], 0.0f);  // relu
  }
}

// final layer: relu fused into row-wise max-pool accumulation
template<int CIN, int FOUT>
__device__ __forceinline__ void mlp_layer_pool(const bf16_t* in, const bf16_t* __restrict__ wt,
                                               float* runmax, int lane) {
  #pragma unroll
  for (int nt = 0; nt < FOUT/16; ++nt) {
    v8f c;
    #pragma unroll
    for (int r = 0; r < 8; ++r) c[r] = 0.0f;
    #pragma unroll
    for (int kt = 0; kt < CIN/32; ++kt) {
      v16bf a = ld_frag_lds(in, kt, lane);
      v16bf w = ld_frag_w<CIN>(wt, nt, kt, lane);
      c = __builtin_amdgcn_wmma_f32_16x16x32_bf16(false, a, false, w, (short)0, c, false, false);
    }
    float m = 0.0f;                               // relu lower bound
    #pragma unroll
    for (int r = 0; r < 8; ++r) m = fmaxf(m, c[r]);   // rows within lane half
    m = fmaxf(m, __shfl_xor(m, 16, 32));          // combine the two lane halves
    runmax[nt] = fmaxf(runmax[nt], m);
  }
}

// ---------------------------------------------------------------------------
// Kernel 4: fused gather + 3-layer MLP + max-pool. One wave32 per query.
// ---------------------------------------------------------------------------
template<int K, int F0, int F1, int F2, int COLOFF>
__global__ __launch_bounds__(128)
void sa_mlp_kernel(const float* __restrict__ xyz, const float* __restrict__ pts,
                   float* out, const int* __restrict__ idxArr,
                   const bf16_t* __restrict__ w0, const bf16_t* __restrict__ w1,
                   const bf16_t* __restrict__ w2) {
  __shared__ alignas(16) bf16_t acts[4][2][16*128];   // ping-pong per wave, 32 KiB
  const int wv   = threadIdx.x >> 5;
  const int lane = threadIdx.x & 31;
  const int q    = blockIdx.x * 4 + wv;               // global query id
  const int b    = q >> 10;
  bf16_t* bufA = acts[wv][0];
  bf16_t* bufB = acts[wv][1];

  if (threadIdx.x == 0) {
    __builtin_prefetch(w0, 0, 0);
    __builtin_prefetch(w1, 0, 0);
    __builtin_prefetch(w2, 0, 0);
  }

  const float* nq = out + OUT_XYZ + (size_t)q*3;
  const float qx = nq[0], qy = nq[1], qz = nq[2];

  float runmax[F2/16];
  #pragma unroll
  for (int i = 0; i < F2/16; ++i) runmax[i] = 0.0f;

  const int row = lane & 15, half = lane >> 4;
  #pragma unroll 1
  for (int mt = 0; mt < K/16; ++mt) {
    // gather 16 rows x 19 channels (points + relative xyz), zero-pad to 32
    int gi = idxArr[(size_t)q*K + mt*16 + row];
    const float* pp = pts + (size_t)(b*N_ + gi)*CPTS;
    const float* pz = xyz + (size_t)(b*N_ + gi)*3;
    #pragma unroll
    for (int cc = 0; cc < 16; ++cc) {
      int col = half*16 + cc;
      float v;
      if      (col < CPTS) v = pp[col];
      else if (col == 16)  v = pz[0] - qx;
      else if (col == 17)  v = pz[1] - qy;
      else if (col == 18)  v = pz[2] - qz;
      else                 v = 0.0f;
      bufA[row*128 + col] = (bf16_t)v;
    }
    asm volatile("s_wait_dscnt 0" ::: "memory");   // per-wave LDS visibility
    mlp_layer<32, F0>(bufA, w0, bufB, lane);
    asm volatile("s_wait_dscnt 0" ::: "memory");
    mlp_layer<F0, F1>(bufB, w1, bufA, lane);
    asm volatile("s_wait_dscnt 0" ::: "memory");
    mlp_layer_pool<F1, F2>(bufA, w2, runmax, lane);
  }

  if (lane < 16) {
    #pragma unroll
    for (int nt = 0; nt < F2/16; ++nt)
      out[OUT_FEAT + (size_t)q*FEAT_TOT + COLOFF + nt*16 + lane] = runmax[nt];
  }
}

// ---------------------------------------------------------------------------
extern "C" void kernel_launch(void* const* d_in, const int* in_sizes, int n_in,
                              void* d_out, int out_size, void* d_ws, size_t ws_size,
                              hipStream_t stream) {
  (void)in_sizes; (void)n_in; (void)out_size; (void)ws_size;
  const float* xyz = (const float*)d_in[0];
  const float* pts = (const float*)d_in[1];
  const float* W[9];
  for (int i = 0; i < 9; ++i) W[i] = (const float*)d_in[2 + i];

  float* out = (float*)d_out;
  char*  ws  = (char*)d_ws;
  int*    i0 = (int*)(ws + IDX0_OFF);
  int*    i1 = (int*)(ws + IDX1_OFF);
  int*    i2 = (int*)(ws + IDX2_OFF);
  bf16_t* wt = (bf16_t*)(ws + WT_OFF);

  // bf16 weight tile offsets (elements) + dims
  const int woff[9] = {0, 1024, 2048,   4096, 6144, 10240,   18432, 20480, 26624};
  const int wcin[9] = {19, 32, 32,      19, 64, 64,          19, 64, 96};
  const int wF[9]   = {32, 32, 64,      64, 64, 128,         64, 96, 128};
  const int wcp[9]  = {32, 32, 32,      32, 64, 64,          32, 64, 96};

  fps_kernel<<<B_, 256, 0, stream>>>(xyz, out);
  qball_kernel<<<B_*4, 256, 0, stream>>>(xyz, out, i0, i1, i2);
  for (int i = 0; i < 9; ++i) {
    int n = wF[i] * wcp[i];
    prep_w_kernel<<<(n + 255)/256, 256, 0, stream>>>(W[i], wt + woff[i],
                                                     wcin[i], wF[i], wcp[i]);
  }
  const int nblk = (B_ * S_) / 4;  // 4 waves (queries) per block
  sa_mlp_kernel<16, 32, 32, 64, 0 ><<<nblk, 128, 0, stream>>>(
      xyz, pts, out, i0, wt + woff[0], wt + woff[1], wt + woff[2]);
  sa_mlp_kernel<32, 64, 64, 128, 64><<<nblk, 128, 0, stream>>>(
      xyz, pts, out, i1, wt + woff[3], wt + woff[4], wt + woff[5]);
  sa_mlp_kernel<64, 64, 96, 128, 192><<<nblk, 128, 0, stream>>>(
      xyz, pts, out, i2, wt + woff[6], wt + woff[7], wt + woff[8]);
}
